// ExpertLayer_16887811408015
// MI455X (gfx1250) — matrix operand order repaired and checked by reading, not compiled, not run on hardware
//
#include <hip/hip_runtime.h>
#include <hip/hip_bf16.h>
#include <stdint.h>

// Problem sizes (match reference)
#define NB 2
#define NS 2048
#define ND 512      // model dim D
#define NH 2048     // hidden dim H
#define NE 16       // experts
#define NT (NB*NS)  // tokens = 4096
#define NA (NT*2)   // assignments (top-2) = 8192

// GEMM tiling
#define BM 128
#define BN 128
#define BK 32
#define LDSTR 40    // padded LDS stride (shorts): 80B, multiple of 16B, bank-friendly

typedef __attribute__((ext_vector_type(16))) __bf16 v16bf;
typedef __attribute__((ext_vector_type(8)))  float  v8f;
typedef __attribute__((ext_vector_type(4)))  unsigned int u32x4;
struct U32x8 { u32x4 lo, hi; };

__device__ __forceinline__ unsigned short f2bf(float f) {
  unsigned int u = __builtin_bit_cast(unsigned int, f);
  u += 0x7fffu + ((u >> 16) & 1u);      // round-to-nearest-even
  return (unsigned short)(u >> 16);
}

// LDS byte offset (addrspace(3) pointer value) for async-to-LDS destination operand.
// Two-step integer cast keeps the host-side pass (64-bit as(3) pointers) legal.
__device__ __forceinline__ unsigned to_lds_off(void* p) {
  return (unsigned)(unsigned long long)(__attribute__((address_space(3))) unsigned short*)p;
}

// CDNA5 async copy: global -> LDS, 128 bits per lane, tracked by ASYNCcnt
__device__ __forceinline__ void async_g2l_b128(void* lds_dst, const void* gsrc) {
  unsigned l = to_lds_off(lds_dst);
  asm volatile("global_load_async_to_lds_b128 %0, %1, off"
               :: "v"(l), "v"(gsrc)
               : "memory");
}
__device__ __forceinline__ void wait_async0() {
  asm volatile("s_wait_asynccnt 0" ::: "memory");
}

// A fragment (16x32 bf16): lane holds 16 contiguous K {k0..k0+15} -> two adjacent 16B chunks
__device__ __forceinline__ v16bf fragA_ld(const unsigned short* p) {
  U32x8 u;
  u.lo = *(const u32x4*)(p);
  u.hi = *(const u32x4*)(p + 8);
  return __builtin_bit_cast(v16bf, u);
}
// B fragment (32x16 bf16, stored transposed [N][K]): lane holds 16 contiguous K of one column
__device__ __forceinline__ v16bf fragB_ld(const unsigned short* p) {
  U32x8 u;
  u.lo = *(const u32x4*)(p);
  u.hi = *(const u32x4*)(p + 8);
  return __builtin_bit_cast(v16bf, u);
}

// ---------------- init: residual accumulator = x, x -> bf16, zero counts --------------
__global__ __launch_bounds__(256) void init_kernel(const float* __restrict__ x,
                                                   float* __restrict__ acc,
                                                   unsigned short* __restrict__ xb,
                                                   int* __restrict__ counts) {
  if (blockIdx.x == 0 && threadIdx.x < NE) counts[threadIdx.x] = 0;
  size_t n = (size_t)NT * ND;
  for (size_t i = (size_t)blockIdx.x * blockDim.x + threadIdx.x; i < n;
       i += (size_t)gridDim.x * blockDim.x) {
    float v = x[i];
    acc[i] = v;
    xb[i] = f2bf(v);
  }
}

// ---------------- weight convert + transpose: fp32 [E][K][N] -> bf16 [E][N][K] ---------
__global__ __launch_bounds__(256) void wconv_kernel(const float* __restrict__ src,
                                                    unsigned short* __restrict__ dst,
                                                    int K, int N) {
  __shared__ unsigned short tile[32][33];
  int e = blockIdx.z;
  int k0 = blockIdx.y * 32;
  int n0 = blockIdx.x * 32;
  const float* s = src + (size_t)e * K * N;
  unsigned short* d = dst + (size_t)e * N * K;
  int t = threadIdx.x;
  #pragma unroll
  for (int p = 0; p < 4; ++p) {
    int idx = p * 256 + t;           // 1024 elements, n fastest -> coalesced reads
    int k = idx >> 5, n = idx & 31;
    tile[k][n] = f2bf(s[(size_t)(k0 + k) * N + n0 + n]);
  }
  __syncthreads();
  #pragma unroll
  for (int p = 0; p < 2; ++p) {
    int idx = p * 256 + t;           // 512 packed u32 writes, k fastest -> coalesced
    int n = idx >> 4, kp = (idx & 15) * 2;
    unsigned lo = tile[kp][n], hi = tile[kp + 1][n];
    *(unsigned*)(&d[(size_t)(n0 + n) * K + k0 + kp]) = lo | (hi << 16);
  }
}

// ---------------- gating: logits, softmax, top-2, atomic ranks -------------------------
__global__ __launch_bounds__(256) void gate_kernel(const float* __restrict__ x,
                                                   const float* __restrict__ Wg,
                                                   const float* __restrict__ bg,
                                                   int* __restrict__ counts,
                                                   int* __restrict__ tok_e,
                                                   int* __restrict__ tok_r,
                                                   float* __restrict__ tok_w) {
  __shared__ float lg[16][17];
  int t = threadIdx.x;
  int lt = t >> 4, e = t & 15;       // 16 tokens x 16 experts per block
  int tok = blockIdx.x * 16 + lt;
  const float* xr = x + (size_t)tok * ND;
  float a = 0.f;
  for (int d = 0; d < ND; ++d) a += xr[d] * Wg[d * NE + e];
  lg[lt][e] = a + bg[e];
  __syncthreads();
  if (e == 0) {
    float m = -1e30f;
    for (int j = 0; j < NE; ++j) m = fmaxf(m, lg[lt][j]);
    float p[NE]; float s = 0.f;
    for (int j = 0; j < NE; ++j) { p[j] = __expf(lg[lt][j] - m); s += p[j]; }
    float inv = 1.f / s;
    int i0 = 0; float v0 = -1.f;
    for (int j = 0; j < NE; ++j) { float pv = p[j] * inv; if (pv > v0) { v0 = pv; i0 = j; } }
    int i1 = 0; float v1 = -1.f;
    for (int j = 0; j < NE; ++j) { if (j == i0) continue; float pv = p[j] * inv; if (pv > v1) { v1 = pv; i1 = j; } }
    int r0 = atomicAdd(&counts[i0], 1);
    int r1 = atomicAdd(&counts[i1], 1);
    tok_e[tok * 2 + 0] = i0; tok_e[tok * 2 + 1] = i1;
    tok_r[tok * 2 + 0] = r0; tok_r[tok * 2 + 1] = r1;
    tok_w[tok * 2 + 0] = v0; tok_w[tok * 2 + 1] = v1;
  }
}

// ---------------- exclusive prefix scan over 16 expert counts --------------------------
__global__ void scan_kernel(const int* __restrict__ counts, int* __restrict__ offs) {
  if (threadIdx.x == 0) {
    int s = 0;
    for (int e = 0; e < NE; ++e) { offs[e] = s; s += counts[e]; }
  }
}

// ---------------- scatter token ids / weights into per-expert contiguous lists ---------
__global__ __launch_bounds__(256) void scatter_kernel(const int* __restrict__ offs,
                                                      const int* __restrict__ tok_e,
                                                      const int* __restrict__ tok_r,
                                                      const float* __restrict__ tok_w,
                                                      int* __restrict__ ltok,
                                                      float* __restrict__ lw) {
  int t = blockIdx.x * blockDim.x + threadIdx.x;
  if (t >= NT) return;
  #pragma unroll
  for (int k = 0; k < 2; ++k) {
    int e = tok_e[t * 2 + k];
    int row = offs[e] + tok_r[t * 2 + k];
    ltok[row] = t;
    lw[row] = tok_w[t * 2 + k];
  }
}

// ---------------- FFN GEMM 1: h = relu(x_gathered @ W1[e] + b1[e]) ---------------------
// A: xb (bf16, gathered rows), B: W1T (bf16, [e][n][k], k contiguous)
__global__ __launch_bounds__(256) void ffn1_kernel(const unsigned short* __restrict__ xb,
                                                   const unsigned short* __restrict__ W1T,
                                                   const float* __restrict__ b1,
                                                   const int* __restrict__ counts,
                                                   const int* __restrict__ offs,
                                                   const int* __restrict__ ltok,
                                                   unsigned short* __restrict__ h) {
  int e = blockIdx.z;
  int count = counts[e];
  int m0 = blockIdx.y * BM;
  if (m0 >= count) return;
  int n0 = blockIdx.x * BN;
  int base = offs[e];
  int tid = threadIdx.x;

  __shared__ __align__(16) unsigned short Asm[2][BM * LDSTR];
  __shared__ __align__(16) unsigned short Bsm[2][BN * LDSTR];
  __shared__ int toks[BM];

  if (tid < BM) {
    int m = m0 + tid;
    toks[tid] = ltok[base + ((m < count) ? m : (count - 1))];
  }
  __syncthreads();

  const unsigned short* Wt = W1T + (size_t)e * NH * ND;   // [n][k], k-len = ND

  int lane = tid & 31, wave = tid >> 5;
  int wm = (wave >> 1) * 32;        // wave M origin within block tile
  int wn = (wave & 1) * 64;         // wave N origin
  int lh = lane >> 4;
  int l15 = lane & 15;
  int r = tid >> 1, half = tid & 1; // staging row / 16B-half per thread

  v8f accv[2][4] = {};

  auto stageA = [&](int buf, int kk) {
    async_g2l_b128(&Asm[buf][r * LDSTR + half * 16],
                   xb + (size_t)toks[r] * ND + kk + half * 16);
  };
  auto stageB = [&](int buf, int kk) {
    async_g2l_b128(&Bsm[buf][r * LDSTR + half * 16],
                   Wt + (size_t)(n0 + r) * ND + kk + half * 16);
  };

  stageA(0, 0); stageB(0, 0);
  const int KC = ND / BK;   // 16
  for (int kc = 0; kc < KC; ++kc) {
    wait_async0();              // own wave's async copies into buf[kc&1] done
    __syncthreads();            // everyone's copies visible
    if (kc + 1 < KC) { stageA((kc + 1) & 1, (kc + 1) * BK); stageB((kc + 1) & 1, (kc + 1) * BK); }
    const unsigned short* Ab = &Asm[kc & 1][0];
    const unsigned short* Bp = &Bsm[kc & 1][0];
    v16bf afr[2], bfr[4];
    int ka = lh * 16;
    #pragma unroll
    for (int i = 0; i < 2; ++i)
      afr[i] = fragA_ld(Ab + (wm + i * 16 + l15) * LDSTR + ka);
    int kb = lh * 16;
    #pragma unroll
    for (int j = 0; j < 4; ++j)
      bfr[j] = fragB_ld(Bp + (wn + j * 16 + l15) * LDSTR + kb);
    #pragma unroll
    for (int i = 0; i < 2; ++i)
      #pragma unroll
      for (int j = 0; j < 4; ++j)
        accv[i][j] = __builtin_amdgcn_wmma_f32_16x16x32_bf16(
            false, afr[i], false, bfr[j], (short)0, accv[i][j], false, false);
  }

  // epilogue: bias + relu -> bf16 hidden
  #pragma unroll
  for (int j = 0; j < 4; ++j) {
    int n = n0 + wn + j * 16 + l15;
    float bias = b1[e * NH + n];
    #pragma unroll
    for (int i = 0; i < 2; ++i) {
      #pragma unroll
      for (int rr = 0; rr < 8; ++rr) {
        int m = m0 + wm + i * 16 + rr + lh * 8;
        if (m < count) {
          float v = accv[i][j][rr] + bias;
          h[(size_t)(base + m) * NH + n] = f2bf(fmaxf(v, 0.f));
        }
      }
    }
  }
}

// ---------------- FFN GEMM 2: y = h @ W2[e] + b2[e]; acc[tok] += w * y -----------------
// A: h (bf16, contiguous rows), B: W2T (bf16, [e][n][k], k contiguous, k-len = NH)
__global__ __launch_bounds__(256) void ffn2_kernel(const unsigned short* __restrict__ h,
                                                   const unsigned short* __restrict__ W2T,
                                                   const float* __restrict__ b2,
                                                   const int* __restrict__ counts,
                                                   const int* __restrict__ offs,
                                                   const int* __restrict__ ltok,
                                                   const float* __restrict__ lw,
                                                   float* __restrict__ out_acc) {
  int e = blockIdx.z;
  int count = counts[e];
  int m0 = blockIdx.y * BM;
  if (m0 >= count) return;
  int n0 = blockIdx.x * BN;
  int base = offs[e];
  int tid = threadIdx.x;

  __shared__ __align__(16) unsigned short Asm[2][BM * LDSTR];
  __shared__ __align__(16) unsigned short Bsm[2][BN * LDSTR];
  __shared__ int toks[BM];
  __shared__ float tws[BM];

  if (tid < BM) {
    int m = m0 + tid;
    int row = base + ((m < count) ? m : (count - 1));
    toks[tid] = ltok[row];
    tws[tid] = (m < count) ? lw[row] : 0.f;
  }
  __syncthreads();

  const unsigned short* Wt = W2T + (size_t)e * ND * NH;

  int lane = tid & 31, wave = tid >> 5;
  int wm = (wave >> 1) * 32;
  int wn = (wave & 1) * 64;
  int lh = lane >> 4;
  int l15 = lane & 15;
  int r = tid >> 1, half = tid & 1;
  int arow = m0 + r; if (arow >= count) arow = count - 1;   // clamp A row

  v8f accv[2][4] = {};

  auto stageA = [&](int buf, int kk) {
    async_g2l_b128(&Asm[buf][r * LDSTR + half * 16],
                   h + (size_t)(base + arow) * NH + kk + half * 16);
  };
  auto stageB = [&](int buf, int kk) {
    async_g2l_b128(&Bsm[buf][r * LDSTR + half * 16],
                   Wt + (size_t)(n0 + r) * NH + kk + half * 16);
  };

  stageA(0, 0); stageB(0, 0);
  const int KC = NH / BK;   // 64
  for (int kc = 0; kc < KC; ++kc) {
    wait_async0();
    __syncthreads();
    if (kc + 1 < KC) { stageA((kc + 1) & 1, (kc + 1) * BK); stageB((kc + 1) & 1, (kc + 1) * BK); }
    const unsigned short* Ab = &Asm[kc & 1][0];
    const unsigned short* Bp = &Bsm[kc & 1][0];
    v16bf afr[2], bfr[4];
    int ka = lh * 16;
    #pragma unroll
    for (int i = 0; i < 2; ++i)
      afr[i] = fragA_ld(Ab + (wm + i * 16 + l15) * LDSTR + ka);
    int kb = lh * 16;
    #pragma unroll
    for (int j = 0; j < 4; ++j)
      bfr[j] = fragB_ld(Bp + (wn + j * 16 + l15) * LDSTR + kb);
    #pragma unroll
    for (int i = 0; i < 2; ++i)
      #pragma unroll
      for (int j = 0; j < 4; ++j)
        accv[i][j] = __builtin_amdgcn_wmma_f32_16x16x32_bf16(
            false, afr[i], false, bfr[j], (short)0, accv[i][j], false, false);
  }

  // epilogue: bias, scale by routing weight, atomic-accumulate into residual buffer
  #pragma unroll
  for (int j = 0; j < 4; ++j) {
    int n = n0 + wn + j * 16 + l15;
    float bias = b2[e * ND + n];
    #pragma unroll
    for (int i = 0; i < 2; ++i) {
      #pragma unroll
      for (int rr = 0; rr < 8; ++rr) {
        int mloc = wm + i * 16 + rr + lh * 8;
        int m = m0 + mloc;
        if (m < count) {
          float y = accv[i][j][rr] + bias;
          atomicAdd(&out_acc[(size_t)toks[mloc] * ND + n], tws[mloc] * y);
        }
      }
    }
  }
}

// ---------------- LayerNorm over D per token ------------------------------------------
__global__ __launch_bounds__(256) void ln_kernel(const float* __restrict__ acc,
                                                 const float* __restrict__ gamma,
                                                 const float* __restrict__ beta,
                                                 float* __restrict__ out) {
  int tok = blockIdx.x;
  const float* row = acc + (size_t)tok * ND;
  int t = threadIdx.x;
  float v0 = row[t], v1 = row[t + 256];
  float s = v0 + v1;
  float q = v0 * v0 + v1 * v1;
  #pragma unroll
  for (int off = 16; off > 0; off >>= 1) {
    s += __shfl_down(s, off, 32);
    q += __shfl_down(q, off, 32);
  }
  __shared__ float ss[8], qq[8];
  __shared__ float mu_s, rs_s;
  int wv = t >> 5;
  if ((t & 31) == 0) { ss[wv] = s; qq[wv] = q; }
  __syncthreads();
  if (t == 0) {
    float S = 0.f, Q = 0.f;
    #pragma unroll
    for (int i = 0; i < 8; ++i) { S += ss[i]; Q += qq[i]; }
    float mu = S * (1.f / ND);
    float var = Q * (1.f / ND) - mu * mu;
    mu_s = mu;
    rs_s = rsqrtf(var + 1e-5f);
  }
  __syncthreads();
  float mu = mu_s, rs = rs_s;
  out[(size_t)tok * ND + t]       = (v0 - mu) * rs * gamma[t]       + beta[t];
  out[(size_t)tok * ND + t + 256] = (v1 - mu) * rs * gamma[t + 256] + beta[t + 256];
}

// ---------------- launch ---------------------------------------------------------------
extern "C" void kernel_launch(void* const* d_in, const int* in_sizes, int n_in,
                              void* d_out, int out_size, void* d_ws, size_t ws_size,
                              hipStream_t stream) {
  const float* x    = (const float*)d_in[0];
  const float* Wg   = (const float*)d_in[1];
  const float* bg   = (const float*)d_in[2];
  const float* W1   = (const float*)d_in[3];
  const float* b1   = (const float*)d_in[4];
  const float* W2   = (const float*)d_in[5];
  const float* b2   = (const float*)d_in[6];
  const float* gam  = (const float*)d_in[7];
  const float* bet  = (const float*)d_in[8];
  float* out = (float*)d_out;

  char* w = (char*)d_ws;
  unsigned short* xb  = (unsigned short*)w;  w += (size_t)NT * ND * 2;   //  4 MB
  unsigned short* h   = (unsigned short*)w;  w += (size_t)NA * NH * 2;   // 32 MB
  unsigned short* W1T = (unsigned short*)w;  w += (size_t)NE * ND * NH * 2; // 32 MB
  unsigned short* W2T = (unsigned short*)w;  w += (size_t)NE * NH * ND * 2; // 32 MB
  float* acc   = (float*)w;                  w += (size_t)NT * ND * 4;   //  8 MB
  int*   counts= (int*)w;                    w += 64;
  int*   offs  = (int*)w;                    w += 64;
  int*   tok_e = (int*)w;                    w += (size_t)NT * 2 * 4;
  int*   tok_r = (int*)w;                    w += (size_t)NT * 2 * 4;
  float* tok_w = (float*)w;                  w += (size_t)NT * 2 * 4;
  int*   ltok  = (int*)w;                    w += (size_t)NA * 4;
  float* lw    = (float*)w;                  w += (size_t)NA * 4;

  init_kernel<<<2048, 256, 0, stream>>>(x, acc, xb, counts);
  // W1 fp32 [E][D][H] -> W1T bf16 [E][H][D];  W2 fp32 [E][H][D] -> W2T bf16 [E][D][H]
  wconv_kernel<<<dim3(NH / 32, ND / 32, NE), 256, 0, stream>>>(W1, W1T, ND, NH);
  wconv_kernel<<<dim3(ND / 32, NH / 32, NE), 256, 0, stream>>>(W2, W2T, NH, ND);
  gate_kernel<<<NT / 16, 256, 0, stream>>>(x, Wg, bg, counts, tok_e, tok_r, tok_w);
  scan_kernel<<<1, 32, 0, stream>>>(counts, offs);
  scatter_kernel<<<NT / 256, 256, 0, stream>>>(offs, tok_e, tok_r, tok_w, ltok, lw);
  ffn1_kernel<<<dim3(NH / BN, NT / BM, NE), 256, 0, stream>>>(xb, W1T, b1, counts, offs, ltok, h);
  ffn2_kernel<<<dim3(ND / BN, NT / BM, NE), 256, 0, stream>>>(h, W2T, b2, counts, offs, ltok, lw, acc);
  ln_kernel<<<NT, 256, 0, stream>>>(acc, gam, bet, out);
}
